// ContextCluster_41979010351423
// MI455X (gfx1250) — matrix operand order repaired and checked by reading, not compiled, not run on hardware
//
#include <hip/hip_runtime.h>

// ---------------------------------------------------------------------------
// ContextCluster forward for MI455X (gfx1250, wave32, WMMA).
//   prep    : Wf/Wv/Wp fp32 -> bf16 (once, into ws)
//   conv_fv : f = Wf*x+bf, v = Wv*x+bv   (bf16 WMMA, fp32 accum)
//   cluster : per-region pooling / cosine / hard-assign / agg (LDS atomics)
//   proj    : out = Wp*mid+bp            (bf16 WMMA, TDM tensor-load staging)
// ---------------------------------------------------------------------------

typedef __attribute__((ext_vector_type(2)))  __bf16 v2bf;
typedef __attribute__((ext_vector_type(8)))  __bf16 v8bf;
typedef __attribute__((ext_vector_type(16))) __bf16 v16bf;
typedef __attribute__((ext_vector_type(8)))  float  v8f;
typedef int          v4i __attribute__((ext_vector_type(4)));
typedef int          v8i __attribute__((ext_vector_type(8)));
typedef unsigned int v4u __attribute__((ext_vector_type(4)));

#define BATCH 16
#define DIM   96
#define HD    256          // HEADS * HEAD_DIM
#define NPIX  4096         // 64*64

#define WF_ELEMS  (HD * DIM)       // 24576
#define WFV_ELEMS (2 * WF_ELEMS)   // 49152 (Wf rows then Wv rows)
#define W16_ELEMS (WFV_ELEMS + DIM * HD)  // + Wp = 73728

// Async-copy pointer casts: global src = AS1 int4*, LDS dst = AS3 int4*.
#define ASYNC_GPTR(p) ((__attribute__((address_space(1))) v4i*)(p))
#define ASYNC_LPTR(p) ((__attribute__((address_space(3))) v4i*)(p))

// ---------------------------------------------------------------------------
// Kernel 0: weight conversion fp32 -> bf16.
// Layout: [0,24576) Wf, [24576,49152) Wv, [49152,73728) Wp (all row-major).
// ---------------------------------------------------------------------------
__global__ __launch_bounds__(256) void prep_weights(
    const float* __restrict__ Wf, const float* __restrict__ Wv,
    const float* __restrict__ Wp, __bf16* __restrict__ W16)
{
  const int idx = blockIdx.x * 256 + threadIdx.x;
  if (idx < WF_ELEMS)            W16[idx] = (__bf16)Wf[idx];
  else if (idx < WFV_ELEMS)      W16[idx] = (__bf16)Wv[idx - WF_ELEMS];
  else if (idx < W16_ELEMS)      W16[idx] = (__bf16)Wp[idx - WFV_ELEMS];
}

// ---------------------------------------------------------------------------
// Kernel 1: f/v 1x1 conv.  Output fv_t[b][pixel][o], o in [0,512):
//   o<256 -> f (Wf,bf), o>=256 -> v (Wv,bv).   Channel-contiguous layout.
// Block = 4 waves; wave owns one 16-row o-tile over a 64-pixel strip.
// LDS x-strip staged K-contiguous: xs[n][k].  Staging is 8-row x float4
// tiles: b128 global loads -> b128 LDS stores.
// ---------------------------------------------------------------------------
__global__ __launch_bounds__(128) void conv_fv_kernel(
    const float* __restrict__ x, const __bf16* __restrict__ W16,
    const float* __restrict__ bf_, const float* __restrict__ bv,
    __bf16* __restrict__ fvt)
{
  __shared__ __bf16 xs[64 * DIM];                    // [n][k], 12 KB
  const int b     = blockIdx.z;
  const int pBase = blockIdx.x * 64;
  const int wave  = threadIdx.x >> 5;
  const int lane  = threadIdx.x & 31;
  const int oTile = (blockIdx.y * 4 + wave) * 16;    // 0..496

  // Stage x[b, :, strip] -> LDS transposed bf16.  (12 k-octs) x (16 n-quads).
  for (int idx = threadIdx.x; idx < 12 * 16; idx += 128) {
    const int k8 = idx >> 4, n4 = idx & 15;
    const float* src = x + ((size_t)b * DIM + k8 * 8) * NPIX + pBase + n4 * 4;
    float4 rows[8];
    #pragma unroll
    for (int r = 0; r < 8; ++r)
      rows[r] = *(const float4*)(src + (size_t)r * NPIX);
    #pragma unroll
    for (int i = 0; i < 4; ++i) {
      v8bf o;
      #pragma unroll
      for (int r = 0; r < 8; ++r)
        o[r] = (__bf16)(((const float*)&rows[r])[i]);
      *(v8bf*)&xs[(n4 * 4 + i) * DIM + k8 * 8] = o;
    }
    __builtin_prefetch(src + 64, 0, 1);              // global_prefetch_b8
  }
  __syncthreads();

  const float* bsel = (oTile < HD) ? bf_ : bv;
  const int    oLoc = (oTile < HD) ? oTile : (oTile - HD);

  const int m  = lane & 15;                          // A row / B,D col
  const int hi = lane >> 4;                          // upper half-wave
  const __bf16* wrow = W16 + (size_t)(oTile + m) * DIM;  // Wf/Wv contiguous

  v8f acc[4] = {};
  #pragma unroll
  for (int chunk = 0; chunk < 3; ++chunk) {          // K = 96 = 3 * 32
    const int kc = chunk * 32;
    // A fragment: 16 contiguous k (k = 16*hi + s), matching B.
    const v16bf a  = *(const v16bf*)(wrow + kc + 16 * hi);
    #pragma unroll
    for (int nt = 0; nt < 4; ++nt) {
      // B fragment: 16 contiguous k at column (nt*16+m) -> one 32B LDS load.
      const v16bf bb =
          *(const v16bf*)&xs[(nt * 16 + m) * DIM + kc + hi * 16];
      acc[nt] = __builtin_amdgcn_wmma_f32_16x16x32_bf16(
          false, a, false, bb, (short)0, acc[nt], false, false);
    }
  }

  // D layout: VGPR r -> row oTile + r + 8*hi, col = m.  Rows are contiguous
  // in fv_t's o dimension -> one packed v8bf (b128) store per n-tile.
  #pragma unroll
  for (int nt = 0; nt < 4; ++nt) {
    v8bf o;
    #pragma unroll
    for (int r = 0; r < 8; ++r)
      o[r] = (__bf16)(acc[nt][r] + bsel[oLoc + hi * 8 + r]);
    const size_t off =
        ((size_t)b * NPIX + pBase + nt * 16 + m) * 512 + oTile + hi * 8;
    *(v8bf*)(fvt + off) = o;
  }
}

// ---------------------------------------------------------------------------
// Kernel 2: clustering.  One 64-thread block per region (8192 regions).
// thread t = point n (0..63); channel-contiguous b128 loads from fv_t.
// ---------------------------------------------------------------------------
__global__ __launch_bounds__(64) void cluster_kernel(
    const __bf16* __restrict__ fvt, const float* __restrict__ alpha_p,
    const float* __restrict__ beta_p, __bf16* __restrict__ midt)
{
  __shared__ float sfT[64][36];     // f points, [point][channel], padded
  __shared__ float svT[64][36];     // v points
  __shared__ float scent[4][32];    // L2-normalized f centers
  __shared__ float svcent[4][32];   // pooled v centers
  __shared__ float sagg[4][32];     // gathered values
  __shared__ float scount[4];       // points per center

  const int region = blockIdx.x;
  const int bh = region >> 6;       // batch*8 + head
  const int fold = region & 63;
  const int f1 = fold >> 3, f2 = fold & 7;
  const int b = bh >> 3, head = bh & 7;
  const int t = threadIdx.x;
  const int w2 = t >> 3, h2 = t & 7;
  const int p = (f1 * 8 + w2) * 64 + (f2 * 8 + h2);

  const float alpha = alpha_p[0], beta = beta_p[0];

  // Load this point's f and v channel vectors: 4+4 b128 loads.
  const __bf16* fptr = fvt + ((size_t)b * NPIX + p) * 512 + head * 32;
  float fvec[32], vvec[32];
  #pragma unroll
  for (int q = 0; q < 4; ++q) {
    const v8bf fq = *(const v8bf*)(fptr + q * 8);
    const v8bf vq = *(const v8bf*)(fptr + HD + q * 8);
    #pragma unroll
    for (int r = 0; r < 8; ++r) {
      fvec[q * 8 + r] = (float)fq[r];
      vvec[q * 8 + r] = (float)vq[r];
    }
  }
  #pragma unroll
  for (int c = 0; c < 32; c += 4) {
    *(float4*)&sfT[t][c] = make_float4(fvec[c], fvec[c+1], fvec[c+2], fvec[c+3]);
    *(float4*)&svT[t][c] = make_float4(vvec[c], vvec[c+1], vvec[c+2], vvec[c+3]);
  }
  for (int idx = t; idx < 128; idx += 64) ((float*)sagg)[idx] = 0.0f;
  if (t < 4) scount[t] = 0.0f;
  __syncthreads();

  // AdaptiveMaxPool2d((2,2)): center m = max over its 4x4 block of points.
  if (t < 4) {                                   // f centers + L2 normalize
    const int pw = t >> 1, ph = t & 1;
    float cf[32]; float nrm = 0.f;
    for (int c = 0; c < 32; ++c) {
      float mv = -3.4e38f;
      for (int i = 0; i < 16; ++i) {
        const int n = (pw * 4 + (i >> 2)) * 8 + ph * 4 + (i & 3);
        mv = fmaxf(mv, sfT[n][c]);
      }
      cf[c] = mv; nrm += mv * mv;
    }
    nrm = fmaxf(sqrtf(nrm), 1e-12f);
    for (int c = 0; c < 32; ++c) scent[t][c] = cf[c] / nrm;
  } else if (t < 8) {                            // v centers
    const int mm = t - 4, pw = mm >> 1, ph = mm & 1;
    for (int c = 0; c < 32; ++c) {
      float mv = -3.4e38f;
      for (int i = 0; i < 16; ++i) {
        const int n = (pw * 4 + (i >> 2)) * 8 + ph * 4 + (i & 3);
        mv = fmaxf(mv, svT[n][c]);
      }
      svcent[mm][c] = mv;
    }
  }
  __syncthreads();

  // Cosine vs 4 centers, sigmoid, hard argmax (first max wins).
  float nrm = 0.f;
  #pragma unroll
  for (int c = 0; c < 32; ++c) nrm += fvec[c] * fvec[c];
  nrm = fmaxf(sqrtf(nrm), 1e-12f);
  float best = -1.0f; int mbest = 0;
  #pragma unroll
  for (int mm = 0; mm < 4; ++mm) {
    float d = 0.f;
    #pragma unroll
    for (int c = 0; c < 32; c += 4) {
      const float4 s4 = *(const float4*)&scent[mm][c];
      d += s4.x * fvec[c] + s4.y * fvec[c+1] + s4.z * fvec[c+2] + s4.w * fvec[c+3];
    }
    const float s = 1.0f / (1.0f + __expf(-(beta + alpha * (d / nrm))));
    if (s > best) { best = s; mbest = mm; }
  }

  // Gather points -> centers (ds_add_f32 LDS atomics).
  atomicAdd(&scount[mbest], 1.0f);
  #pragma unroll
  for (int c = 0; c < 32; ++c)
    atomicAdd(&sagg[mbest][c], vvec[c] * best);
  __syncthreads();

  if (t < 4) {
    const float inv = 1.0f / (scount[t] + 1.0f);
    for (int c = 0; c < 32; ++c)
      sagg[t][c] = (sagg[t][c] + svcent[t][c]) * inv;
  }
  __syncthreads();

  // Dispatch centers -> points; mid_t[b][pixel][c] channel-contiguous.
  const size_t obase = ((size_t)b * NPIX + p) * HD + head * 32;
  #pragma unroll
  for (int q = 0; q < 4; ++q) {
    v8bf o;
    #pragma unroll
    for (int r = 0; r < 8; ++r)
      o[r] = (__bf16)(sagg[mbest][q * 8 + r] * best);
    *(v8bf*)(midt + obase + q * 8) = o;
  }
}

// ---------------------------------------------------------------------------
// Kernel 3: out = Wp * mid + bp  (96 x 256 x 4096 per batch).
// mid_t strip (64 pixels x 256 ch) is CONTIGUOUS in memory and equals the
// LDS layout we need -> single TDM TENSOR_LOAD_TO_LDS per workgroup
// (TENSORcnt), falling back to per-lane GLOBAL_LOAD_ASYNC_TO_LDS_B128.
// Block = 6 waves (o-tiles 0..80); 32 KB LDS strip.
// ---------------------------------------------------------------------------
__global__ __launch_bounds__(192) void proj_kernel(
    const __bf16* __restrict__ midt, const __bf16* __restrict__ Wp16,
    const float* __restrict__ bp, float* __restrict__ out)
{
  __shared__ __bf16 ms[64 * HD];                     // [n][k], 32 KB
  const int b     = blockIdx.z;
  const int pBase = blockIdx.x * 64;
  const int wave  = threadIdx.x >> 5;
  const int lane  = threadIdx.x & 31;
  const int oTile = wave * 16;

  const __bf16* gsrc = midt + ((size_t)b * NPIX + pBase) * HD;

#if __has_builtin(__builtin_amdgcn_tensor_load_to_lds)
  if (threadIdx.x == 0) {
    // D# group 0 (ISA 8.3): count=1, lds_addr, global_addr, type=2.
    const unsigned long long ga = (unsigned long long)(size_t)gsrc;
    const unsigned ldsA =
        (unsigned)(unsigned long long)(__attribute__((address_space(3))) void*)ms;
    v4u g0;
    g0[0] = 1u;                                      // count=1
    g0[1] = ldsA;                                    // lds_addr
    g0[2] = (unsigned)(ga & 0xffffffffu);            // global_addr[31:0]
    g0[3] = (unsigned)((ga >> 32) & 0x01ffffffu)     // global_addr[56:32]
          | (2u << 30);                              // type=2 ("image")
    // D# group 1 (ISA 8.4): 1-D copy of 16384 bf16 (32 KB).
    v8i g1;
    g1[0] = (int)(1u << 16);          // workgroup_mask=0, data_size=1 (2B)
    g1[1] = (int)(0x4000u << 16);     // tensor_dim0 = 16384 (low 16 bits)
    g1[2] = (int)(1u << 16);          // tensor_dim0 hi=0 | tensor_dim1 = 1
    g1[3] = (int)(0x4000u << 16);     // tensor_dim1 hi=0 | tile_dim0 = 16384
    g1[4] = 1;                        // tile_dim1 = 1, tile_dim2 = 0
    g1[5] = 16384;                    // tensor_dim0_stride (low 32)
    g1[6] = (int)(0x4000u << 16);     // stride0 hi=0 | tensor_dim1_stride lo
    g1[7] = 0;                        // tensor_dim1_stride hi
    v4i z4 = {0, 0, 0, 0};
#if __clang_major__ >= 23
    v8i z8 = {0, 0, 0, 0, 0, 0, 0, 0};
    __builtin_amdgcn_tensor_load_to_lds(g0, g1, z4, z4, z8, 0);
#else
    __builtin_amdgcn_tensor_load_to_lds(g0, g1, z4, z4, 0);
#endif
#if __has_builtin(__builtin_amdgcn_s_wait_tensorcnt)
    __builtin_amdgcn_s_wait_tensorcnt(0);
#else
    asm volatile("s_wait_tensorcnt 0x0" ::: "memory");
#endif
  }
#elif __has_builtin(__builtin_amdgcn_global_load_async_to_lds_b128)
  for (int idx = threadIdx.x; idx < 64 * (HD / 8); idx += 192) {
    const int n = idx >> 5, c8 = idx & 31;           // 32 x 16B chunks / pixel
    __builtin_amdgcn_global_load_async_to_lds_b128(
        ASYNC_GPTR(gsrc + (size_t)n * HD + c8 * 8),
        ASYNC_LPTR(&ms[n * HD + c8 * 8]), 0, 0);
  }
#if __has_builtin(__builtin_amdgcn_s_wait_asynccnt)
  __builtin_amdgcn_s_wait_asynccnt(0);
#else
  asm volatile("s_wait_asynccnt 0x0" ::: "memory");
#endif
#else
  for (int idx = threadIdx.x; idx < 64 * (HD / 8); idx += 192) {
    const int n = idx >> 5, c8 = idx & 31;
    *(v8bf*)&ms[n * HD + c8 * 8] = *(const v8bf*)(gsrc + (size_t)n * HD + c8 * 8);
  }
#endif
  __syncthreads();

  const int m  = lane & 15;
  const int hi = lane >> 4;
  const __bf16* wrow = Wp16 + (size_t)(oTile + m) * HD;

  v8f acc[4] = {};
  #pragma unroll
  for (int chunk = 0; chunk < 8; ++chunk) {          // K = 256 = 8 * 32
    const int kc = chunk * 32;
    const v16bf a  = *(const v16bf*)(wrow + kc + 16 * hi);
    #pragma unroll
    for (int nt = 0; nt < 4; ++nt) {
      const v16bf bb =
          *(const v16bf*)&ms[(nt * 16 + m) * HD + kc + hi * 16];
      acc[nt] = __builtin_amdgcn_wmma_f32_16x16x32_bf16(
          false, a, false, bb, (short)0, acc[nt], false, false);
    }
  }

  #pragma unroll
  for (int nt = 0; nt < 4; ++nt) {
    #pragma unroll
    for (int r = 0; r < 8; ++r) {
      const int rloc = r + hi * 8;
      out[((size_t)b * DIM + oTile + rloc) * NPIX + pBase + nt * 16 + m] =
          acc[nt][r] + bp[oTile + rloc];
    }
  }
}

// ---------------------------------------------------------------------------
extern "C" void kernel_launch(void* const* d_in, const int* in_sizes, int n_in,
                              void* d_out, int out_size, void* d_ws, size_t ws_size,
                              hipStream_t stream) {
  const float* x     = (const float*)d_in[0];
  const float* Wf    = (const float*)d_in[1];
  const float* bf_   = (const float*)d_in[2];
  const float* Wv    = (const float*)d_in[3];
  const float* bv    = (const float*)d_in[4];
  const float* Wp    = (const float*)d_in[5];
  const float* bp    = (const float*)d_in[6];
  const float* alpha = (const float*)d_in[7];
  const float* beta  = (const float*)d_in[8];

  // Workspace layout (bytes):
  //   W16   @ 0         : 73728 bf16  (147456 B)
  //   fv_t  @ 147456    : 16*4096*512 bf16 (64 MiB)
  //   mid_t @ +64 MiB   : 16*4096*256 bf16 (32 MiB)
  __bf16* W16  = (__bf16*)d_ws;
  __bf16* fvt  = (__bf16*)((char*)d_ws + 147456);
  __bf16* midt = (__bf16*)((char*)d_ws + 147456 +
                           (size_t)BATCH * NPIX * 512 * sizeof(__bf16));
  __bf16* Wp16 = W16 + WFV_ELEMS;

  prep_weights<<<dim3((W16_ELEMS + 255) / 256), 256, 0, stream>>>(Wf, Wv, Wp, W16);
  conv_fv_kernel<<<dim3(64, 8, BATCH), 128, 0, stream>>>(x, W16, bf_, bv, fvt);
  cluster_kernel<<<dim3(8192), 64, 0, stream>>>(fvt, alpha, beta, midt);
  proj_kernel<<<dim3(64, 1, BATCH), 192, 0, stream>>>(midt, Wp16, bp, (float*)d_out);
}
